// MultiHeadCrossAttention_11330123727037
// MI455X (gfx1250) — compile-verified, matrix operand order checked by reading, not run on hardware
//
#include <hip/hip_runtime.h>

// ---------------------------------------------------------------------------
// MultiHeadCrossAttention for MI455X (gfx1250), wave32 + WMMA bf16.
// B=8, NQ=1024, NK=2048, EMB=1024, H=16, D=64.
// Pipeline: (1) weight transpose->bf16, (2) bf16 WMMA GEMMs for Q/K/V,
// (3) fused flash-attention (online softmax), (4) bf16 WMMA GEMM -> f32 out.
// K-chunks of 64 (16 WMMA per barrier pair), packed bf16 LDS stores, and
// gfx1250 async global->LDS copies (ASYNCcnt) where the builtin exists.
// ---------------------------------------------------------------------------

typedef __bf16 bf16;
typedef __attribute__((ext_vector_type(16))) __bf16 v16bf;
typedef __attribute__((ext_vector_type(8)))  float  v8f;
typedef __attribute__((ext_vector_type(4)))  int    v4i_t;

union FragBF { v16bf v; unsigned int u[8]; };
union AccF   { v8f   v; float f[8]; };
union PackBF { bf16 h[2]; unsigned int u; };

#define Bv   8
#define NQv  1024
#define NKv  2048
#define EMBv 1024
#define Hv   16
#define Dv   64

// ---- gfx1250 async global->LDS copy (ASYNCcnt), with safe fallback --------
#if defined(__has_builtin)
#if __has_builtin(__builtin_amdgcn_global_load_async_to_lds_b128) && \
    __has_builtin(__builtin_amdgcn_s_wait_asynccnt)
#define HAVE_ASYNC_LDS 1
#endif
#endif

#ifdef HAVE_ASYNC_LDS
typedef __attribute__((address_space(1))) v4i_t as1_v4i;
typedef __attribute__((address_space(3))) v4i_t as3_v4i;
__device__ inline void async_copy16(const bf16* g, bf16* l) {
    __builtin_amdgcn_global_load_async_to_lds_b128(
        (as1_v4i*)(void*)g, (as3_v4i*)(void*)l, 0, 0);
}
__device__ inline void async_wait0() { __builtin_amdgcn_s_wait_asynccnt(0); }
#else
__device__ inline void async_copy16(const bf16* g, bf16* l) {
    *(uint4*)l = *(const uint4*)g;
}
__device__ inline void async_wait0() {}
#endif

// Load a 16x32 bf16 WMMA operand fragment from row-major storage.
// Per ISA 7.12.2 (16-bit A 16x32): lane L holds row L%16; VGPR v holds the
// K-pair at kbase = (v<4 ? 2v : 16+2(v-4)) + 8*(L/16).
__device__ inline v16bf load_frag(const bf16* base, int stride, int lane) {
    const int row = lane & 15;
    const int hf  = lane >> 4;
    const bf16* p = base + row * stride;
    FragBF f;
#pragma unroll
    for (int v = 0; v < 8; ++v) {
        const int kb = (v < 4) ? (v * 2 + hf * 8) : (16 + (v - 4) * 2 + hf * 8);
        f.u[v] = *(const unsigned int*)(p + kb);
    }
    return f.v;
}

// ---------------------------------------------------------------------------
// Kernel 1: W[k][n] (f32) -> Wt[n][k] (bf16), 1024x1024
// ---------------------------------------------------------------------------
__global__ __launch_bounds__(256) void transpose_cvt_kernel(
    const float* __restrict__ W, bf16* __restrict__ Wt) {
    const int idx = blockIdx.x * 256 + threadIdx.x;   // 0..1048575
    const int n = idx >> 10;
    const int k = idx & 1023;
    Wt[(size_t)n * 1024 + k] = (bf16)W[(size_t)k * 1024 + n];
}

// ---------------------------------------------------------------------------
// Kernel 2: Y = X(f32)[M][K] @ Wt(bf16,N-major)[N][K] + bias, via WMMA bf16.
// Block = 256 threads (8 waves), 128x128 tile, K-chunks of 64 (2 wmma steps).
// Wave w: rows 32*(w&3), cols 64*(w>>2): 2x4 accumulators of 16x16.
// ---------------------------------------------------------------------------
#define GBM 128
#define GBN 128
#define GBK 64
#define GSTR (GBK + 8)   // 72 elems = 144B rows: 16B-aligned chunks, bank-spread

template<bool OUT_FP32>
__global__ __launch_bounds__(256) void gemm_xw_kernel(
    const float* __restrict__ X, const bf16* __restrict__ Wt,
    const float* __restrict__ bias, void* __restrict__ Y,
    int M, int N, int K) {
    __shared__ bf16 As[GBM][GSTR];
    __shared__ bf16 Bs[GBN][GSTR];

    const int tid  = threadIdx.x;
    const int lane = tid & 31;
    const int wave = tid >> 5;
    const int m0 = blockIdx.y * GBM;
    const int n0 = blockIdx.x * GBN;
    const int wm = (wave & 3) * 32;
    const int wn = (wave >> 2) * 64;

    AccF acc[2][4];
#pragma unroll
    for (int i = 0; i < 2; ++i)
#pragma unroll
        for (int j = 0; j < 4; ++j)
#pragma unroll
            for (int e = 0; e < 8; ++e) acc[i][j].f[e] = 0.0f;

    for (int k0 = 0; k0 < K; k0 += GBK) {
        __syncthreads();
        // B tile: 128x64 bf16 straight copy -> async to LDS (16B per lane-op).
#pragma unroll
        for (int i = 0; i < 4; ++i) {
            const int idx = tid + i * 256;       // 0..1023 chunks of 8 elems
            const int row = idx >> 3;
            const int c8  = (idx & 7) * 8;
            async_copy16(Wt + (size_t)(n0 + row) * K + k0 + c8, &Bs[row][c8]);
        }
        // A tile: 128x64 f32 -> bf16 convert, packed b32 LDS stores.
#pragma unroll
        for (int i = 0; i < 8; ++i) {
            const int idx = tid + i * 256;       // 0..2047
            const int row = idx >> 4;            // 16 float4 per row
            const int kc  = (idx & 15) * 4;
            const float4 xv = *(const float4*)(X + (size_t)(m0 + row) * K + k0 + kc);
            PackBF p0, p1;
            p0.h[0] = (bf16)xv.x; p0.h[1] = (bf16)xv.y;
            p1.h[0] = (bf16)xv.z; p1.h[1] = (bf16)xv.w;
            *(unsigned int*)&As[row][kc + 0] = p0.u;
            *(unsigned int*)&As[row][kc + 2] = p1.u;
        }
        async_wait0();
        __syncthreads();

#pragma unroll
        for (int c = 0; c < 2; ++c) {
            v16bf af[2], bf[4];
#pragma unroll
            for (int i = 0; i < 2; ++i)
                af[i] = load_frag(&As[wm + i * 16][c * 32], GSTR, lane);
#pragma unroll
            for (int j = 0; j < 4; ++j)
                bf[j] = load_frag(&Bs[wn + j * 16][c * 32], GSTR, lane);
#pragma unroll
            for (int i = 0; i < 2; ++i)
#pragma unroll
                for (int j = 0; j < 4; ++j)
                    acc[i][j].v = __builtin_amdgcn_wmma_f32_16x16x32_bf16(
                        false, af[i], false, bf[j], (short)0, acc[i][j].v,
                        false, false);
        }
    }

    // Epilogue. C layout: lane holds col = lane&15, VGPR r = row r + 8*(lane>>4).
    const int col16 = lane & 15;
    const int hf    = lane >> 4;
#pragma unroll
    for (int i = 0; i < 2; ++i) {
#pragma unroll
        for (int j = 0; j < 4; ++j) {
            const int coln = n0 + wn + j * 16 + col16;
            const float bv2 = bias[coln];
#pragma unroll
            for (int r = 0; r < 8; ++r) {
                const int rowm = m0 + wm + i * 16 + r + hf * 8;
                const float val = acc[i][j].f[r] + bv2;
                if (OUT_FP32) ((float*)Y)[(size_t)rowm * N + coln] = val;
                else          ((bf16*)Y)[(size_t)rowm * N + coln] = (bf16)val;
            }
        }
    }
}

// ---------------------------------------------------------------------------
// Kernel 3: flash attention. Grid (NQ/128, H, B); block 256 (8 waves);
// wave w owns q-rows [q0+16w, q0+16w+16). K/V streamed in 64-wide chunks.
// scale = 1/sqrt(EMB) = 1/32 (matches reference: scales by sqrt(emb_size)).
// ---------------------------------------------------------------------------
#define BKV  64
#define KSTR (Dv + 8)    // 72: Ks rows (d-contig), 144B -> 16B-aligned chunks
#define VSTR (BKV + 8)   // 72: Vt / Pst rows (kv-contig)

__global__ __launch_bounds__(256) void flash_attn_kernel(
    const bf16* __restrict__ Qb, const bf16* __restrict__ Kb,
    const bf16* __restrict__ Vb, float* __restrict__ Ob) {
    __shared__ bf16 Ks[BKV][KSTR];           // K chunk, row = kv, col = d
    __shared__ bf16 Vt[Dv][VSTR];            // V chunk transposed: row = d
    __shared__ bf16 Pst[8][16][VSTR];        // per-wave P staging (A-major)

    const int tid  = threadIdx.x;
    const int lane = tid & 31;
    const int wave = tid >> 5;
    const int h = blockIdx.y;
    const int b = blockIdx.z;
    const int q0 = blockIdx.x * 128 + wave * 16;
    const int col16 = lane & 15;
    const int hf    = lane >> 4;

    // Q fragments for this wave's 16 rows, kept in registers (D=64 -> 2 frags)
    v16bf qf[2];
    {
        const bf16* qbase = Qb + (size_t)(b * NQv + q0) * EMBv + h * Dv;
        qf[0] = load_frag(qbase,      EMBv, lane);
        qf[1] = load_frag(qbase + 32, EMBv, lane);
    }

    float mrow[8], lrow[8];
    AccF oacc[4];
#pragma unroll
    for (int r = 0; r < 8; ++r) { mrow[r] = -1e30f; lrow[r] = 0.0f; }
#pragma unroll
    for (int dt = 0; dt < 4; ++dt)
#pragma unroll
        for (int e = 0; e < 8; ++e) oacc[dt].f[e] = 0.0f;

    for (int kv0 = 0; kv0 < NKv; kv0 += BKV) {
        __syncthreads();  // protect prev-iteration LDS consumers
        // K chunk: 64 x 64 bf16 = 8KB, async 16B copies (2 per thread).
#pragma unroll
        for (int i = 0; i < 2; ++i) {
            const int idx = tid + i * 256;   // 0..511 chunks of 8 elems
            const int row = idx >> 3;        // kv
            const int c8  = (idx & 7) * 8;   // d
            async_copy16(Kb + (size_t)(b * NKv + kv0 + row) * EMBv + h * Dv + c8,
                         &Ks[row][c8]);
        }
        // V chunk, transposed into LDS: Vt[d][kv] = V[kv][d]
#pragma unroll
        for (int i = 0; i < 16; ++i) {
            const int idx = tid + i * 256;   // 0..4095
            const int row = idx >> 6;        // kv
            const int d   = idx & 63;
            Vt[d][row] = Vb[(size_t)(b * NKv + kv0 + row) * EMBv + h * Dv + d];
        }
        async_wait0();
        __syncthreads();

        // S = Q @ K^T : four 16x16 col tiles, K-dim = D = 64 (2 wmma steps)
        AccF s[4];
#pragma unroll
        for (int t = 0; t < 4; ++t) {
#pragma unroll
            for (int e = 0; e < 8; ++e) s[t].f[e] = 0.0f;
#pragma unroll
            for (int c = 0; c < 2; ++c) {
                const v16bf kf = load_frag(&Ks[t * 16][c * 32], KSTR, lane);
                s[t].v = __builtin_amdgcn_wmma_f32_16x16x32_bf16(
                    false, qf[c], false, kf, (short)0, s[t].v, false, false);
            }
        }

        // Online softmax. Row R = r + 8*hf lives in VGPR r across the lane's
        // 16-lane half; shfl_xor(8,4,2,1) stays within the half.
#pragma unroll
        for (int r = 0; r < 8; ++r) {
            float sv[4];
#pragma unroll
            for (int t = 0; t < 4; ++t) sv[t] = s[t].f[r] * 0.03125f;
            float cm = fmaxf(fmaxf(sv[0], sv[1]), fmaxf(sv[2], sv[3]));
#pragma unroll
            for (int off = 8; off >= 1; off >>= 1)
                cm = fmaxf(cm, __shfl_xor(cm, off, 32));
            const float mnew = fmaxf(mrow[r], cm);
            float pv[4];
            float rs = 0.0f;
#pragma unroll
            for (int t = 0; t < 4; ++t) { pv[t] = __expf(sv[t] - mnew); rs += pv[t]; }
#pragma unroll
            for (int off = 8; off >= 1; off >>= 1)
                rs += __shfl_xor(rs, off, 32);
            const float alpha = __expf(mrow[r] - mnew);
            lrow[r] = lrow[r] * alpha + rs;
            mrow[r] = mnew;
#pragma unroll
            for (int dt = 0; dt < 4; ++dt) oacc[dt].f[r] *= alpha;
            // C-layout -> A-layout relayout through per-wave LDS staging
#pragma unroll
            for (int t = 0; t < 4; ++t)
                Pst[wave][r + hf * 8][t * 16 + col16] = (bf16)pv[t];
        }
        __syncthreads();

        // O += P @ V  (A = P 16x64, B rows = Vt rows = output d columns)
#pragma unroll
        for (int c = 0; c < 2; ++c) {
            const v16bf pf = load_frag(&Pst[wave][0][c * 32], VSTR, lane);
#pragma unroll
            for (int dt = 0; dt < 4; ++dt) {
                const v16bf vf = load_frag(&Vt[dt * 16][c * 32], VSTR, lane);
                oacc[dt].v = __builtin_amdgcn_wmma_f32_16x16x32_bf16(
                    false, pf, false, vf, (short)0, oacc[dt].v, false, false);
            }
        }
    }

    // Normalize and write O (f32, [B][NQ][EMB], head-concat layout).
#pragma unroll
    for (int r = 0; r < 8; ++r) {
        const float inv = 1.0f / lrow[r];
        const int qrow = q0 + r + hf * 8;
#pragma unroll
        for (int dt = 0; dt < 4; ++dt)
            Ob[(size_t)(b * NQv + qrow) * EMBv + h * Dv + dt * 16 + col16] =
                oacc[dt].f[r] * inv;
    }
}

// ---------------------------------------------------------------------------
// Host-side orchestration.
// Workspace layout (needs 120 MB):
//   [0,2M)    Wq^T bf16      [2M,4M)   Wk^T bf16
//   [4M,6M)   Wv^T bf16      [6M,8M)   Wp^T bf16
//   [8M,24M)  Q bf16         [24M,56M) K bf16
//   [56M,88M) V bf16         [88M,120M) O f32
// ---------------------------------------------------------------------------
extern "C" void kernel_launch(void* const* d_in, const int* in_sizes, int n_in,
                              void* d_out, int out_size, void* d_ws, size_t ws_size,
                              hipStream_t stream) {
    const float* qtk = (const float*)d_in[0];
    const float* img = (const float*)d_in[1];
    const float* Wq  = (const float*)d_in[2];
    const float* bq  = (const float*)d_in[3];
    const float* Wk  = (const float*)d_in[4];
    const float* bk  = (const float*)d_in[5];
    const float* Wv  = (const float*)d_in[6];
    const float* bv  = (const float*)d_in[7];
    const float* Wp  = (const float*)d_in[8];
    const float* bp  = (const float*)d_in[9];
    float* out = (float*)d_out;

    char* ws = (char*)d_ws;
    const size_t MB = 1024u * 1024u;
    bf16*  Wqt  = (bf16*)(ws + 0 * MB);
    bf16*  Wkt  = (bf16*)(ws + 2 * MB);
    bf16*  Wvt  = (bf16*)(ws + 4 * MB);
    bf16*  Wpt  = (bf16*)(ws + 6 * MB);
    bf16*  Qbuf = (bf16*)(ws + 8 * MB);
    bf16*  Kbuf = (bf16*)(ws + 24 * MB);
    bf16*  Vbuf = (bf16*)(ws + 56 * MB);
    float* Obuf = (float*)(ws + 88 * MB);

    // 1) weight transpose+convert
    transpose_cvt_kernel<<<4096, 256, 0, stream>>>(Wq, Wqt);
    transpose_cvt_kernel<<<4096, 256, 0, stream>>>(Wk, Wkt);
    transpose_cvt_kernel<<<4096, 256, 0, stream>>>(Wv, Wvt);
    transpose_cvt_kernel<<<4096, 256, 0, stream>>>(Wp, Wpt);

    // 2) projections (bf16 out)
    gemm_xw_kernel<false><<<dim3(8, 64),  256, 0, stream>>>(qtk, Wqt, bq, Qbuf, Bv * NQv, EMBv, EMBv);
    gemm_xw_kernel<false><<<dim3(8, 128), 256, 0, stream>>>(img, Wkt, bk, Kbuf, Bv * NKv, EMBv, EMBv);
    gemm_xw_kernel<false><<<dim3(8, 128), 256, 0, stream>>>(img, Wvt, bv, Vbuf, Bv * NKv, EMBv, EMBv);

    // 3) fused flash attention
    flash_attn_kernel<<<dim3(NQv / 128, Hv, Bv), 256, 0, stream>>>(Qbuf, Kbuf, Vbuf, Obuf);

    // 4) output projection (f32 out)
    gemm_xw_kernel<true><<<dim3(8, 64), 256, 0, stream>>>(Obuf, Wpt, bp, out, Bv * NQv, EMBv, EMBv);
}
